// LSTM_46875273068878
// MI455X (gfx1250) — compile-verified, matrix operand order checked by reading, not run on hardware
//
#include <hip/hip_runtime.h>
#include <hip/hip_bf16.h>

// ---------------------------------------------------------------------------
// Persistent fused LSTM for gfx1250 (MI455X).
//   T=2048, B=32, IN=H=512, gate dim G=4H=2048.
//   32 workgroups x 256 threads (8 wave32). WG `wg` owns h-columns
//   [wg*16, wg*16+16) and the 4 corresponding gate row-blocks of W_ih/W_hh,
//   resident in LDS as bf16 for the entire sequence.
//   Per step: gates = X_t @ Wih_blk^T + Hprev @ Whh_blk^T + bias (WMMA bf16),
//   elementwise LSTM cell, h exchanged device-wide via a bf16 double buffer
//   (staged into LDS with global_load_async_to_lds_b128) + per-step atomic
//   grid barrier. x_{t+1} is converted into a second LDS buffer during step t
//   so the f32->bf16 VALU work co-executes with the WMMA pipe.
// ---------------------------------------------------------------------------

#define T_STEPS 2048
#define BATCH   32
#define IN_DIM  512
#define H_DIM   512
#define NWG     32          // workgroups; each owns H_DIM/NWG = 16 h-columns
#define NCOL    16          // h-columns per WG
#define NTHREADS 256        // 8 wave32

typedef __attribute__((ext_vector_type(16))) __bf16 bf16x16;
typedef __attribute__((ext_vector_type(8)))  float  f32x8;

// LDS partition (bytes):
//   Wih 64*512 bf16 | Whh 64*512 bf16 | X double buffer 2*32*512 bf16 |
//   H 32*512 bf16 | gates 4*32*16 f32 | c 32*16 f32 | bias 64 f32
#define SMEM_BYTES (64*512*2 + 64*512*2 + 2*32*512*2 + 32*512*2 \
                    + 4*32*16*4 + 32*16*4 + 64*4)

__device__ __forceinline__ unsigned short f2bf(float f) {
    unsigned u = __float_as_uint(f);
    unsigned r = (u >> 16) & 1u;
    return (unsigned short)((u + 0x7fffu + r) >> 16);
}

__device__ __forceinline__ float fsigmoid(float x) {
    return 1.0f / (1.0f + __expf(-x));
}
__device__ __forceinline__ float ftanh(float x) {
    return 1.0f - 2.0f / (__expf(2.0f * x) + 1.0f);   // tanh(x)
}

// 16x32 bf16 fragment load (A-layout) from an LDS tile, row stride ldk elems.
__device__ __forceinline__ bf16x16 load_frag(const unsigned short* tile, int row0,
                                             int ldk, int kk, int lane) {
    const int half = (lane >> 4) & 1;
    const int r    = lane & 15;
    const unsigned short* p = tile + (row0 + r) * ldk + kk + half * 8;
    union { bf16x16 v; uint4 u[2]; } tmp;
    tmp.u[0] = *(const uint4*)(p);        // K = kk + half*8      .. +7
    tmp.u[1] = *(const uint4*)(p + 16);   // K = kk + 16 + half*8 .. +7
    return tmp.v;
}

// f32 -> bf16 conversion of one 32x512 activation slice into an LDS buffer.
__device__ __forceinline__ void stage_x(const float* __restrict__ xt,
                                        unsigned short* __restrict__ xbuf, int tid) {
    for (int i = tid * 4; i < BATCH * IN_DIM; i += NTHREADS * 4) {
        float4 v = *(const float4*)(xt + i);
        xbuf[i + 0] = f2bf(v.x);
        xbuf[i + 1] = f2bf(v.y);
        xbuf[i + 2] = f2bf(v.z);
        xbuf[i + 3] = f2bf(v.w);
    }
}

// Per-step grid barrier using a fresh counter per timestep (zeroed by init kernel).
__device__ __forceinline__ void grid_barrier(unsigned* cnt, int nwg) {
    __syncthreads();
    if (threadIdx.x == 0) {
        __threadfence();                 // make h_buf writes device-visible
        atomicAdd(cnt, 1u);
        while (__hip_atomic_load(cnt, __ATOMIC_RELAXED, __HIP_MEMORY_SCOPE_AGENT)
               < (unsigned)nwg) {
            __builtin_amdgcn_s_sleep(1);
        }
        __threadfence();
    }
    __syncthreads();
}

__global__ void lstm_init_kernel(const float* __restrict__ h0,
                                 unsigned short* __restrict__ h_buf,
                                 unsigned* __restrict__ counters) {
    int i = blockIdx.x * blockDim.x + threadIdx.x;
    if (i < T_STEPS) counters[i] = 0u;
    if (i < BATCH * H_DIM) h_buf[i] = f2bf(h0[i]);
}

__launch_bounds__(NTHREADS, 1)
__global__ void lstm_persistent_kernel(const float* __restrict__ seq,
                                       const float* __restrict__ c0,
                                       const float* __restrict__ W_ih,
                                       const float* __restrict__ W_hh,
                                       const float* __restrict__ b_ih,
                                       const float* __restrict__ b_hh,
                                       float* __restrict__ out,
                                       unsigned short* __restrict__ h_buf,
                                       unsigned* __restrict__ counters) {
    extern __shared__ unsigned char smem_raw[];
    unsigned short* w_ih_s  = (unsigned short*)smem_raw;            // [64][512]
    unsigned short* w_hh_s  = w_ih_s + 64 * 512;                    // [64][512]
    unsigned short* x0_s    = w_hh_s + 64 * 512;                    // [32][512]
    unsigned short* x1_s    = x0_s + 32 * 512;                      // [32][512]
    unsigned short* h_s     = x1_s + 32 * 512;                      // [32][512]
    float*          gates_s = (float*)(h_s + 32 * 512);             // [4][32][16]
    float*          c_s     = gates_s + 4 * 32 * 16;                // [32][16]
    float*          bias_s  = c_s + 32 * 16;                        // [64]

    const int wg   = blockIdx.x;           // 0..31
    const int tid  = threadIdx.x;
    const int lane = tid & 31;
    const int wave = tid >> 5;             // 0..7
    const int mi   = wave & 1;             // batch-tile (2 x 16 rows)
    const int ni   = wave >> 1;            // gate-block (4 x 16 cols) == gate id

    // ---- Preload resident weights (bf16) + bias + c0 slice -----------------
    for (int i = tid; i < 64 * 512; i += NTHREADS) {
        int row  = i >> 9;                 // 0..63 = g*16 + r
        int k    = i & 511;
        int g    = row >> 4;
        int r    = row & 15;
        int grow = g * H_DIM + wg * NCOL + r;   // row in [0, 2048)
        w_ih_s[i] = f2bf(W_ih[(size_t)grow * IN_DIM + k]);
        w_hh_s[i] = f2bf(W_hh[(size_t)grow * H_DIM + k]);
    }
    if (tid < 64) {
        int g = tid >> 4, r = tid & 15;
        int grow = g * H_DIM + wg * NCOL + r;
        bias_s[tid] = b_ih[grow] + b_hh[grow];
    }
    for (int i = tid; i < BATCH * NCOL; i += NTHREADS) {
        int b = i >> 4, col = i & 15;
        c_s[i] = c0[b * H_DIM + wg * NCOL + col];
    }
    // Stage x_0 into buffer 0.
    stage_x(seq, x0_s, tid);

    float* hf = out + (size_t)T_STEPS * BATCH * H_DIM;   // final h
    float* cf = hf + BATCH * H_DIM;                      // final c

    const unsigned h_lds_base = (unsigned)(size_t)h_s;   // LDS byte offset of h tile

    // h_buf[0] was initialized by lstm_init_kernel before this launch.
    grid_barrier(&counters[0], NWG);

    for (int t = 0; t < T_STEPS; ++t) {
        // ---- Async-stage h_prev (bf16) straight into LDS (no VGPR bounce) --
        {
            const unsigned short* hsrc = h_buf + (size_t)(t & 1) * BATCH * H_DIM;
            #pragma unroll
            for (int j = 0; j < (BATCH * H_DIM * 2) / (16 * NTHREADS); ++j) {
                unsigned off  = (unsigned)(tid + j * NTHREADS) * 16u;
                unsigned ldsa = h_lds_base + off;
                const void* g = (const char*)hsrc + off;
                asm volatile("global_load_async_to_lds_b128 %0, %1, off"
                             :: "v"(ldsa), "v"(g) : "memory");
            }
        }
        if (t + 2 < T_STEPS) {
            // warm L2 for the activation slice we convert next step
            __builtin_prefetch(seq + (size_t)(t + 2) * BATCH * IN_DIM + tid * 64, 0, 1);
        }
        asm volatile("s_wait_asynccnt 0x0" ::: "memory");
        __syncthreads();

        const unsigned short* x_cur = (t & 1) ? x1_s : x0_s;
        unsigned short*       x_nxt = (t & 1) ? x0_s : x1_s;

        // ---- WMMA: D[32x64] = X[32x512] Wih^T + H[32x512] Whh^T ------------
        // Software-pipelined: load fragments for iteration i+1 before WMMA i
        // so ds_load latency overlaps matrix issue (partial s_wait_dscnt).
        f32x8 acc = {};
        bf16x16 a_cur = load_frag(x_cur,  mi * 16, 512, 0, lane);
        bf16x16 b_cur = load_frag(w_ih_s, ni * 16, 512, 0, lane);
        #pragma unroll
        for (int i = 0; i < 32; ++i) {
            bf16x16 a_nxt = a_cur, b_nxt = b_cur;
            if (i + 1 < 32) {
                const unsigned short* ta = ((i + 1) < 16) ? x_cur  : h_s;
                const unsigned short* tb = ((i + 1) < 16) ? w_ih_s : w_hh_s;
                const int kk = ((i + 1) & 15) * 32;
                a_nxt = load_frag(ta, mi * 16, 512, kk, lane);
                b_nxt = load_frag(tb, ni * 16, 512, kk, lane);
            }
            acc = __builtin_amdgcn_wmma_f32_16x16x32_bf16(
                      false, a_cur, false, b_cur, (short)0, acc, false, false);
            a_cur = a_nxt;
            b_cur = b_nxt;
        }

        // ---- Convert x_{t+1} into the other LDS buffer (overlaps XDL) ------
        if (t + 1 < T_STEPS) {
            stage_x(seq + (size_t)(t + 1) * BATCH * IN_DIM, x_nxt, tid);
        }

        // ---- Scatter accumulator to gates LDS (C/D layout) + bias ----------
        {
            const int n     = lane & 15;             // gate column within 16
            const int mbase = mi * 16 + ((lane >> 4) << 3);
            const float bia = bias_s[ni * 16 + n];
            #pragma unroll
            for (int r = 0; r < 8; ++r) {
                gates_s[(ni * 32 + (mbase + r)) * 16 + n] = acc[r] + bia;
            }
        }
        __syncthreads();

        // ---- Elementwise LSTM cell for this WG's 16 columns ----------------
        unsigned short* h_next = h_buf + (size_t)((t + 1) & 1) * BATCH * H_DIM;
        for (int e = tid; e < BATCH * NCOL; e += NTHREADS) {
            int b = e >> 4, col = e & 15;
            float gi = gates_s[(0 * 32 + b) * 16 + col];
            float gf = gates_s[(1 * 32 + b) * 16 + col];
            float gg = gates_s[(2 * 32 + b) * 16 + col];
            float go = gates_s[(3 * 32 + b) * 16 + col];
            float i_ = fsigmoid(gi);
            float f_ = fsigmoid(gf);
            float g_ = ftanh(gg);
            float o_ = fsigmoid(go);
            float c_new = f_ * c_s[e] + i_ * g_;
            float h_new = o_ * ftanh(c_new);
            c_s[e] = c_new;
            int gcol = wg * NCOL + col;
            // streaming output: never re-read, keep it out of L2's hot set
            __builtin_nontemporal_store(h_new,
                &out[(size_t)t * BATCH * H_DIM + b * H_DIM + gcol]);
            h_next[b * H_DIM + gcol] = f2bf(h_new);
            if (t == T_STEPS - 1) hf[b * H_DIM + gcol] = h_new;
        }

        if (t + 1 < T_STEPS) grid_barrier(&counters[t + 1], NWG);
    }

    __syncthreads();
    for (int e = tid; e < BATCH * NCOL; e += NTHREADS) {
        int b = e >> 4, col = e & 15;
        cf[b * H_DIM + wg * NCOL + col] = c_s[e];
    }
}

extern "C" void kernel_launch(void* const* d_in, const int* in_sizes, int n_in,
                              void* d_out, int out_size, void* d_ws, size_t ws_size,
                              hipStream_t stream) {
    (void)in_sizes; (void)n_in; (void)out_size; (void)ws_size;
    const float* seq  = (const float*)d_in[0];
    const float* h0   = (const float*)d_in[1];
    const float* c0   = (const float*)d_in[2];
    const float* W_ih = (const float*)d_in[3];
    const float* W_hh = (const float*)d_in[4];
    const float* b_ih = (const float*)d_in[5];
    const float* b_hh = (const float*)d_in[6];
    float* out = (float*)d_out;

    // Workspace layout: [0, 8KB) per-step barrier counters; then bf16 h double buffer.
    unsigned*       counters = (unsigned*)d_ws;
    unsigned short* h_buf    = (unsigned short*)((char*)d_ws + T_STEPS * sizeof(unsigned));

    lstm_init_kernel<<<(BATCH * H_DIM + 255) / 256, 256, 0, stream>>>(h0, h_buf, counters);

    hipFuncSetAttribute(reinterpret_cast<const void*>(lstm_persistent_kernel),
                        hipFuncAttributeMaxDynamicSharedMemorySize, SMEM_BYTES);

    lstm_persistent_kernel<<<NWG, NTHREADS, SMEM_BYTES, stream>>>(
        seq, c0, W_ih, W_hh, b_ih, b_hh, out, h_buf, counters);
}
